// MDLSTM2D_63556926046903
// MI455X (gfx1250) — matrix operand order repaired and checked by reading, not compiled, not run on hardware
//
#include <hip/hip_runtime.h>
#include <hip/hip_bf16.h>

typedef __bf16 bf16_t;
typedef __attribute__((ext_vector_type(16))) __bf16 v16bf;
typedef __attribute__((ext_vector_type(8)))  float  v8f;
typedef int v4i __attribute__((vector_size(16)));

#define B_   16
#define D_   64
#define F_   128
#define U_   128
#define TILE (16 * 128)          // one (i,j) tile: 16 rows (batch) x 128 cols
#define NCELL (D_ * D_)

#if __has_builtin(__builtin_amdgcn_global_load_async_to_lds_b128)
#define HAVE_ASYNC_LDS 1
#endif

__device__ __forceinline__ void wait_asynccnt0() {
#if __has_builtin(__builtin_amdgcn_s_wait_asynccnt)
    __builtin_amdgcn_s_wait_asynccnt(0);
#else
    asm volatile("s_wait_asynccnt 0" ::: "memory");
#endif
}

// ---------------------------------------------------------------------------
// Fragment loaders (CDNA5 WMMA 16-bit layouts, wave32)
// A (16x32 bf16): lanes 0-15 row M=lane, v0-3 = K 0..7, v4-7 = K 16..23
//                 lanes16-31 row M=lane-16, v0-3 = K 8..15, v4-7 = K 24..31
// ---------------------------------------------------------------------------
__device__ __forceinline__ v16bf load_a_frag(const bf16_t* tile, int lane, int kk) {
    const int row = lane & 15;
    const int kb  = kk * 32 + ((lane & 16) ? 8 : 0);
    const bf16_t* p = tile + row * 128 + kb;
    union { uint4 q[2]; v16bf v; } u;
    u.q[0] = *(const uint4*)(p);        // K +0..7
    u.q[1] = *(const uint4*)(p + 16);   // K +16..23
    return u.v;
}

// B (32x16 bf16): lane holds column N=lane&15; lanes 0-15 K 0..7,16..23, lanes 16-31
// K 8..15,24..31, 2 packed per VGPR. Weights stored column-major: ut[n*128 + k].
__device__ __forceinline__ v16bf load_b_frag(const bf16_t* ut, int lane, int nt, int kk) {
    const int n  = nt * 16 + (lane & 15);
    const int kb = kk * 32 + ((lane & 16) ? 8 : 0);
    const bf16_t* p = ut + n * 128 + kb;
    union { uint4 q[2]; v16bf v; } u;
    u.q[0] = *(const uint4*)(p);        // K +0..7
    u.q[1] = *(const uint4*)(p + 16);   // K +16..23
    return u.v;
}

__device__ __forceinline__ v8f wmma_bf16(v16bf a, v16bf b, v8f c) {
    return __builtin_amdgcn_wmma_f32_16x16x32_bf16(
        false, a, false, b, (short)0, c, false, false);
}

__device__ __forceinline__ float sigmoidf_(float x) {
    return 1.0f / (1.0f + __expf(-x));
}

// ---------------------------------------------------------------------------
// prep: x [B,D,D,F] f32  ->  xb [cell][b][f] bf16   (GEMM-A friendly layout)
// ---------------------------------------------------------------------------
__global__ __launch_bounds__(256) void prep_x_kernel(const float* __restrict__ x,
                                                     bf16_t* __restrict__ xb) {
    const int cell = blockIdx.x;                    // i*64 + j
    const int i = cell >> 6, j = cell & 63;
    const int tid = threadIdx.x;
    const int b  = tid >> 4;                        // 16 threads per row
    const int f0 = (tid & 15) * 8;
    const float* src = x + (((size_t)(b * D_ + i) * D_ + j) * F_) + f0;
    bf16_t* dst = xb + (size_t)cell * TILE + b * 128 + f0;
#pragma unroll
    for (int t = 0; t < 8; ++t) dst[t] = (bf16_t)src[t];
}

// ---------------------------------------------------------------------------
// prep: W[g] (f,u) f32 -> wtb[g][u][f] bf16 ; U[g] (k,n) f32 -> utb[g][n][k]
// ---------------------------------------------------------------------------
__global__ __launch_bounds__(256) void prep_w_kernel(
    const float* w0, const float* w1, const float* w2, const float* w3, const float* w4,
    const float* u0, const float* u1, const float* u2, const float* u3, const float* u4,
    bf16_t* __restrict__ wtb, bf16_t* __restrict__ utb) {
    const int idx = blockIdx.x * 256 + threadIdx.x; // < 5*128*128
    const int g   = idx >> 14;
    const int rem = idx & 16383;
    const int n = rem >> 7, k = rem & 127;
    const float* ws[5] = {w0, w1, w2, w3, w4};
    const float* us[5] = {u0, u1, u2, u3, u4};
    wtb[idx] = (bf16_t)ws[g][k * 128 + n];
    utb[idx] = (bf16_t)us[g][k * 128 + n];
}

__global__ __launch_bounds__(256) void init_flags_kernel(int* __restrict__ flags) {
    flags[blockIdx.x * 256 + threadIdx.x] = 0;
}

// ---------------------------------------------------------------------------
// Input projections: P[g][cell][b][u] = x_tile @ W[g] + bias[g]
// One block per cell, 8 waves, wave nt owns N columns [16nt,16nt+16).
// k-outer / gate-inner so the 10 B-loads per k-step clause together and one
// wait covers 5 back-to-back WMMAs.
// ---------------------------------------------------------------------------
__global__ __launch_bounds__(256) void proj_kernel(
    const bf16_t* __restrict__ xb, const bf16_t* __restrict__ wtb,
    const float* b0, const float* b1, const float* b2, const float* b3, const float* b4,
    float* __restrict__ P) {
    const int cell = blockIdx.x;
    const int wv = threadIdx.x >> 5, lane = threadIdx.x & 31;
    const int N = wv * 16 + (lane & 15);
    const int Moff = (lane & 16) ? 8 : 0;
    const bf16_t* At = xb + (size_t)cell * TILE;
    const float* biases[5] = {b0, b1, b2, b3, b4};

    v16bf afr[4];
#pragma unroll
    for (int kk = 0; kk < 4; ++kk) afr[kk] = load_a_frag(At, lane, kk);

    v8f acc[5];
#pragma unroll
    for (int g = 0; g < 5; ++g) {
        const float bv = biases[g][N];
        acc[g] = (v8f){bv, bv, bv, bv, bv, bv, bv, bv};
    }

#pragma unroll
    for (int kk = 0; kk < 4; ++kk) {
        v16bf bfr[5];
#pragma unroll
        for (int g = 0; g < 5; ++g)
            bfr[g] = load_b_frag(wtb + g * 16384, lane, wv, kk);
#pragma unroll
        for (int g = 0; g < 5; ++g)
            acc[g] = wmma_bf16(afr[kk], bfr[g], acc[g]);
    }

#pragma unroll
    for (int g = 0; g < 5; ++g) {
        float* out = P + ((size_t)g * NCELL + cell) * TILE;
#pragma unroll
        for (int r = 0; r < 8; ++r)
            out[(r + Moff) * 128 + N] = acc[g][r];
    }
}

// ---------------------------------------------------------------------------
// Persistent wavefront scan. Block i owns row i, sweeps j = 0..63.
// Up-neighbor handshake via global flags (release/acquire, agent scope).
// All 5 recurrent U matrices (160 KB bf16) live in LDS for the whole sweep.
// Gates: 0=f0 (h_up), 1=f1 (h_left), 2..4 = i/o/c (h_up + h_left).
// ---------------------------------------------------------------------------
__global__ __launch_bounds__(256) void scan_kernel(
    const float* __restrict__ P, const bf16_t* __restrict__ utb,
    bf16_t* __restrict__ Hbf, float* __restrict__ Cst,
    float* __restrict__ out, int* __restrict__ flags) {
    __shared__ bf16_t sU[5 * U_ * U_];                 // 160 KB recurrent weights
    __shared__ bf16_t sUp[TILE];                       // 4 KB
    __shared__ bf16_t sLeft[TILE];                     // 4 KB
    __shared__ bf16_t sSum[TILE];                      // 4 KB

    const int i   = blockIdx.x;
    const int tid = threadIdx.x;
    const int wv = tid >> 5, lane = tid & 31;
    const int N = wv * 16 + (lane & 15);
    const int Moff = (lane & 16) ? 8 : 0;

    // ---- Stage U matrices into LDS (async DMA path when available) ----
    {
        const uint4* src = (const uint4*)utb;          // 10240 x 16B
        uint4* dst = (uint4*)sU;
#if HAVE_ASYNC_LDS
        for (int t = tid; t < 10240; t += 256)
            __builtin_amdgcn_global_load_async_to_lds_b128(
                (__attribute__((address_space(1))) v4i*)(src + t),
                (__attribute__((address_space(3))) v4i*)(dst + t), 0, 0);
        wait_asynccnt0();
#else
        for (int t = tid; t < 10240; t += 256) dst[t] = src[t];
#endif
        // zero the boundary tiles
        const uint4 z = {0u, 0u, 0u, 0u};
        ((uint4*)sUp)[tid]   = z;
        ((uint4*)sLeft)[tid] = z;
    }
    __syncthreads();

    float cprev[8];                                    // c_left lives in VGPRs
#pragma unroll
    for (int r = 0; r < 8; ++r) cprev[r] = 0.0f;

    for (int j = 0; j < D_; ++j) {
        const int cell = i * D_ + j;
        const size_t pcell = (size_t)cell * TILE;

        // ---- accumulators from input projections: independent of the
        //      recurrence, issue BEFORE the neighbor wait to hide latency ----
        v8f acc[5];
#pragma unroll
        for (int g = 0; g < 5; ++g) {
            const float* p = P + (size_t)g * NCELL * TILE + pcell;
#pragma unroll
            for (int r = 0; r < 8; ++r) acc[g][r] = p[(r + Moff) * 128 + N];
        }
        // ---- prefetch next column's projections into cache ----
        if (j + 1 < D_) {
            const size_t pnext = pcell + TILE;
#pragma unroll
            for (int g = 0; g < 5; ++g)
                __builtin_prefetch(P + (size_t)g * NCELL * TILE + pnext + tid * 8, 0, 1);
        }

        // ---- wait for up-neighbor, then stage h_up ----
        if (i > 0) {
            if (tid == 0) {
                while (__hip_atomic_load(&flags[cell - D_], __ATOMIC_ACQUIRE,
                                         __HIP_MEMORY_SCOPE_AGENT) == 0)
                    __builtin_amdgcn_s_sleep(1);
            }
            __syncthreads();
            ((uint4*)sUp)[tid] =
                ((const uint4*)(Hbf + (size_t)(cell - D_) * TILE))[tid];
        }
        __syncthreads();
        {
            const int e0 = tid * 8;
#pragma unroll
            for (int t = 0; t < 8; ++t)
                sSum[e0 + t] = (bf16_t)((float)sUp[e0 + t] + (float)sLeft[e0 + t]);
        }
        __syncthreads();

#pragma unroll
        for (int kk = 0; kk < 4; ++kk) {
            const v16bf aup = load_a_frag(sUp,   lane, kk);
            const v16bf alf = load_a_frag(sLeft, lane, kk);
            const v16bf asm_ = load_a_frag(sSum, lane, kk);
            v16bf bfr[5];
#pragma unroll
            for (int g = 0; g < 5; ++g)
                bfr[g] = load_b_frag(sU + g * 16384, lane, wv, kk);
            acc[0] = wmma_bf16(aup,  bfr[0], acc[0]);
            acc[1] = wmma_bf16(alf,  bfr[1], acc[1]);
            acc[2] = wmma_bf16(asm_, bfr[2], acc[2]);
            acc[3] = wmma_bf16(asm_, bfr[3], acc[3]);
            acc[4] = wmma_bf16(asm_, bfr[4], acc[4]);
        }

        // ---- gate math + state update ----
        float hreg[8];
#pragma unroll
        for (int r = 0; r < 8; ++r) {
            const int M = r + Moff;
            const float f0 = sigmoidf_(acc[0][r]);
            const float f1 = sigmoidf_(acc[1][r]);
            const float it = sigmoidf_(acc[2][r]);
            const float ot = sigmoidf_(acc[3][r]);
            const float cp = tanhf(acc[4][r]);
            const float cu = (i > 0)
                ? Cst[(size_t)(cell - D_) * TILE + M * 128 + N] : 0.0f;
            const float c = f0 * cu + f1 * cprev[r] + it * cp;
            const float h = ot * tanhf(c);
            cprev[r] = c;
            hreg[r]  = h;
            Cst[pcell + M * 128 + N] = c;              // for row i+1 (c_up)
            Hbf[pcell + M * 128 + N] = (bf16_t)h;      // for row i+1 (h_up)
            out[(((size_t)M * D_ + i) * D_ + j) * 128 + N] = h;  // [B,D0,D1,U]
        }

        // ---- recycle h into sLeft for the next column ----
        __syncthreads();                               // WMMA reads of sLeft done
#pragma unroll
        for (int r = 0; r < 8; ++r)
            sLeft[(r + Moff) * 128 + N] = (bf16_t)hreg[r];
        __threadfence();                               // Hbf/Cst visible device-wide
        __syncthreads();
        if (tid == 0)
            __hip_atomic_store(&flags[cell], 1, __ATOMIC_RELEASE,
                               __HIP_MEMORY_SCOPE_AGENT);
    }
}

// ---------------------------------------------------------------------------
extern "C" void kernel_launch(void* const* d_in, const int* in_sizes, int n_in,
                              void* d_out, int out_size, void* d_ws, size_t ws_size,
                              hipStream_t stream) {
    (void)in_sizes; (void)n_in; (void)out_size; (void)ws_size;

    const float* x  = (const float*)d_in[0];
    const float* w[5]   = {(const float*)d_in[1],  (const float*)d_in[4],
                           (const float*)d_in[7],  (const float*)d_in[10],
                           (const float*)d_in[13]};
    const float* u[5]   = {(const float*)d_in[2],  (const float*)d_in[5],
                           (const float*)d_in[8],  (const float*)d_in[11],
                           (const float*)d_in[14]};
    const float* bia[5] = {(const float*)d_in[3],  (const float*)d_in[6],
                           (const float*)d_in[9],  (const float*)d_in[12],
                           (const float*)d_in[15]};

    char* ws = (char*)d_ws;
    bf16_t* xb  = (bf16_t*)ws;  ws += (size_t)NCELL * TILE * 2;        // 16.8 MB
    bf16_t* wtb = (bf16_t*)ws;  ws += (size_t)5 * 128 * 128 * 2;       // 160 KB
    bf16_t* utb = (bf16_t*)ws;  ws += (size_t)5 * 128 * 128 * 2;       // 160 KB
    float*  P   = (float*)ws;   ws += (size_t)5 * NCELL * TILE * 4;    // 167.8 MB
    bf16_t* Hbf = (bf16_t*)ws;  ws += (size_t)NCELL * TILE * 2;        // 16.8 MB
    float*  Cst = (float*)ws;   ws += (size_t)NCELL * TILE * 4;        // 33.6 MB
    int*    flg = (int*)ws;     ws += (size_t)NCELL * 4;               // 16 KB

    prep_x_kernel<<<NCELL, 256, 0, stream>>>(x, xb);
    prep_w_kernel<<<320, 256, 0, stream>>>(w[0], w[1], w[2], w[3], w[4],
                                           u[0], u[1], u[2], u[3], u[4], wtb, utb);
    init_flags_kernel<<<NCELL / 256, 256, 0, stream>>>(flg);
    proj_kernel<<<NCELL, 256, 0, stream>>>(xb, wtb, bia[0], bia[1], bia[2],
                                           bia[3], bia[4], P);
    scan_kernel<<<D_, 256, 0, stream>>>(P, utb, Hbf, Cst, (float*)d_out, flg);
}
